// WindowAttentionV2_53790170415168
// MI455X (gfx1250) — matrix operand order repaired and checked by reading, not compiled, not run on hardware
//
#include <hip/hip_runtime.h>

// ---------------------------------------------------------------------------
// SwinV2 window attention, fully fused per-window kernel for gfx1250 (CDNA5).
// All GEMMs run on v_wmma_f32_16x16x32_bf16 (wave32 WMMA). One 256-thread
// block (8 waves) per window; wave w owns head w in the attention phase.
// ---------------------------------------------------------------------------

typedef __bf16 bf16_t;
typedef bf16_t v16bf __attribute__((ext_vector_type(16)));
typedef float  v8f   __attribute__((ext_vector_type(8)));
typedef unsigned int u32x4 __attribute__((ext_vector_type(4)));

union ABreg { v16bf v; u32x4 q[2]; };

__device__ __forceinline__ unsigned short f2bf(float f) {
    unsigned int u = __builtin_bit_cast(unsigned int, f);
    u += 0x7FFFu + ((u >> 16) & 1u);          // round-to-nearest-even
    return (unsigned short)(u >> 16);
}
__device__ __forceinline__ float bf2f(unsigned short h) {
    unsigned int u = ((unsigned int)h) << 16;
    return __builtin_bit_cast(float, u);
}

// A operand (16x32 bf16), ISA 7.12.2: lane holds row (lane&15);
// K order used here (not the ISA table order): lanes 0-15 carry K 0..15, lanes 16-31 K 16..31.
// `row` points at K=0 of this lane's row (contiguous K in memory).
__device__ __forceinline__ v16bf load_matA(const unsigned short* row, int lane) {
    const int kb = (lane >> 4) << 4;
    ABreg r;
    r.q[0] = *(const u32x4*)(row + kb);
    r.q[1] = *(const u32x4*)(row + kb + 8);
    return r.v;
}
// B operand (32x16 bf16): lane holds column (lane&15);
// lanes 0-15 carry K 0..15, lanes 16-31 carry K 16..31 (contiguous).
// `col` points at K=0 of this lane's column (column stored contiguously).
__device__ __forceinline__ v16bf load_matB(const unsigned short* col, int lane) {
    const int kb = (lane >> 4) << 4;
    ABreg r;
    r.q[0] = *(const u32x4*)(col + kb);
    r.q[1] = *(const u32x4*)(col + kb + 8);
    return r.v;
}
__device__ __forceinline__ v8f wmma_bf16(v16bf a, v16bf b, v8f c) {
    return __builtin_amdgcn_wmma_f32_16x16x32_bf16(false, a, false, b,
                                                   (short)0, c, false, false);
}

// ---------------------------------------------------------------------------
// Prep kernel 1: transpose + convert weights to bf16 (so every WMMA B column
// is a contiguous 16B load). qkvT[768][256], projT[256][256].
// ---------------------------------------------------------------------------
__global__ void convert_weights(const float* __restrict__ qkv_w,
                                const float* __restrict__ proj_w,
                                unsigned short* __restrict__ qkvT,
                                unsigned short* __restrict__ projT) {
    const int o = blockIdx.x;      // out-dim
    const int c = threadIdx.x;     // in-dim (256 threads)
    if (o < 768) qkvT[o * 256 + c] = f2bf(qkv_w[c * 768 + o]);
    else {
        const int oo = o - 768;
        projT[oo * 256 + c] = f2bf(proj_w[c * 256 + oo]);
    }
}

// ---------------------------------------------------------------------------
// Prep kernel 2: continuous-position-bias MLP + gather + 16*sigmoid.
// rel_bias[h][i][j], 8*64*64 f32. ~21 MFLOP total -> trivial.
// ---------------------------------------------------------------------------
__global__ void cpb_bias(const float* __restrict__ coords_table,
                         const float* __restrict__ mlp1_w,
                         const float* __restrict__ mlp1_b,
                         const float* __restrict__ mlp2_w,
                         const int* __restrict__ rel_idx,
                         float* __restrict__ rel_bias) {
    const int i = blockIdx.x;   // 64
    const int j = threadIdx.x;  // 64
    const int idx = rel_idx[i * 64 + j];
    const float t0 = coords_table[idx * 2 + 0];
    const float t1 = coords_table[idx * 2 + 1];
    float acc[8] = {0.f, 0.f, 0.f, 0.f, 0.f, 0.f, 0.f, 0.f};
    for (int hc = 0; hc < 512; ++hc) {
        float hv = fmaxf(t0 * mlp1_w[hc] + t1 * mlp1_w[512 + hc] + mlp1_b[hc], 0.0f);
#pragma unroll
        for (int hd = 0; hd < 8; ++hd) acc[hd] += hv * mlp2_w[hc * 8 + hd];
    }
#pragma unroll
    for (int hd = 0; hd < 8; ++hd)
        rel_bias[hd * 4096 + i * 64 + j] = 16.0f / (1.0f + __expf(-acc[hd]));
}

// ---------------------------------------------------------------------------
// Fused per-window attention kernel. gridDim.x = 2048 windows, 256 threads.
// ---------------------------------------------------------------------------
__global__ __launch_bounds__(256) void win_attn_fused(
    const float* __restrict__ x, const float* __restrict__ mask,
    const unsigned short* __restrict__ qkvT, const float* __restrict__ q_bias,
    const float* __restrict__ v_bias, const float* __restrict__ logit_scale,
    const float* __restrict__ rel_bias, const unsigned short* __restrict__ projT,
    const float* __restrict__ proj_b, float* __restrict__ out)
{
    __shared__ __align__(16) unsigned short sXO[64 * 256];  // X (ph1-2) then O (ph5-6)
    __shared__ __align__(16) unsigned short sQ [64 * 256];  // [tok][h*32+d]
    __shared__ __align__(16) unsigned short sK [64 * 256];
    __shared__ __align__(16) unsigned short sVT[8 * 32 * 64]; // [h][d][tok]
    __shared__ __align__(16) unsigned short sP [8 * 64 * 64]; // [h][i][j]

    const int tid   = threadIdx.x;
    const int lane  = tid & 31;
    const int wv    = tid >> 5;            // wave id == head id
    const int jl    = lane & 15;           // tile column
    const int rhalf = (lane >> 4) << 3;    // C/D row offset (M += 8 for hi half)
    const int blk   = blockIdx.x;

    // ---- Phase 1: stage X [64,256] f32 -> bf16 LDS ----
    {
        const float4* xg = (const float4*)(x + (size_t)blk * (64 * 256));
        for (int e = tid; e < (64 * 256) / 4; e += 256) {
            float4 f = xg[e];
            unsigned long long pk =
                (unsigned long long)f2bf(f.x) |
                ((unsigned long long)f2bf(f.y) << 16) |
                ((unsigned long long)f2bf(f.z) << 32) |
                ((unsigned long long)f2bf(f.w) << 48);
            *(unsigned long long*)(sXO + e * 4) = pk;
        }
    }
    __syncthreads();

    // ---- Phase 2: qkv GEMM [64,256] x [256,768], split into Q/K/V^T ----
    for (int t = wv; t < 192; t += 8) {               // 4 M-tiles x 48 N-tiles
        const int mt = t / 48, nt = t % 48;
        const int jo = nt * 16 + jl;                  // global qkv out-dim
        v8f acc = {};
        const unsigned short* arow = sXO + (mt * 16 + jl) * 256;
        const unsigned short* bcol = qkvT + (size_t)jo * 256;
#pragma unroll
        for (int ks = 0; ks < 8; ++ks)
            acc = wmma_bf16(load_matA(arow + ks * 32, lane),
                            load_matB(bcol + ks * 32, lane), acc);
        const float bias = (jo < 256) ? q_bias[jo]
                         : ((jo < 512) ? 0.0f : v_bias[jo - 512]);
#pragma unroll
        for (int vi = 0; vi < 8; ++vi) {
            const int i = mt * 16 + rhalf + vi;
            const unsigned short h16 = f2bf(acc[vi] + bias);
            if (jo < 256)      sQ[i * 256 + jo] = h16;
            else if (jo < 512) sK[i * 256 + (jo - 256)] = h16;
            else {
                const int hh = (jo - 512) >> 5, dd = (jo - 512) & 31;
                sVT[hh * 2048 + dd * 64 + i] = h16;   // transposed for B operand
            }
        }
    }
    __syncthreads();

    // ---- Phase 3: fold l2-norm (and per-head logit scale) into Q/K rows ----
    // l2norm(q).l2norm(k)*scale == (q*scale/|q|) . (k/|k|)
    for (int r = tid; r < 1024; r += 256) {
        const bool isQ = r < 512;
        const int rr = r & 511, i = rr >> 3, h = rr & 7;
        unsigned short* p = (isQ ? sQ : sK) + i * 256 + h * 32;
        float v[32], ss = 0.0f;
#pragma unroll
        for (int d = 0; d < 32; ++d) { v[d] = bf2f(p[d]); ss += v[d] * v[d]; }
        float f = 1.0f / fmaxf(sqrtf(ss), 1e-12f);
        if (isQ) f *= __expf(fminf(logit_scale[h], 4.6051702f)); // ln(100) clamp
#pragma unroll
        for (int d = 0; d < 32; ++d) p[d] = f2bf(v[d] * f);
    }
    __syncthreads();

    // ---- Phase 4: attention scores + softmax (wave == head) ----
    const int h = wv;
    const float* maskw = mask     + (size_t)(blk & 255) * 4096; // window-in-image
    const float* rbh   = rel_bias + (size_t)h * 4096;
    for (int mt = 0; mt < 4; ++mt) {
        v8f acc[4];
#pragma unroll
        for (int nt = 0; nt < 4; ++nt)
#pragma unroll
            for (int vi = 0; vi < 8; ++vi) {          // C preload = bias + mask
                const int i = mt * 16 + rhalf + vi;
                const int j = nt * 16 + jl;
                acc[nt][vi] = rbh[i * 64 + j] + maskw[i * 64 + j];
            }
        v16bf a = load_matA(sQ + (mt * 16 + jl) * 256 + h * 32, lane); // K = D = 32
#pragma unroll
        for (int nt = 0; nt < 4; ++nt)
            acc[nt] = wmma_bf16(a,
                load_matB(sK + (nt * 16 + jl) * 256 + h * 32, lane), acc[nt]);
        // softmax along j: each (vi, lane-half) is one row spread over 16 lanes
#pragma unroll
        for (int vi = 0; vi < 8; ++vi) {
            float m = fmaxf(fmaxf(acc[0][vi], acc[1][vi]),
                            fmaxf(acc[2][vi], acc[3][vi]));
            for (int s = 1; s <= 8; s <<= 1) m = fmaxf(m, __shfl_xor(m, s));
            float e0 = __expf(acc[0][vi] - m), e1 = __expf(acc[1][vi] - m);
            float e2 = __expf(acc[2][vi] - m), e3 = __expf(acc[3][vi] - m);
            float sum = e0 + e1 + e2 + e3;
            for (int s = 1; s <= 8; s <<= 1) sum += __shfl_xor(sum, s);
            const float rs = 1.0f / sum;
            const int i = mt * 16 + rhalf + vi;
            unsigned short* prow = sP + h * 4096 + i * 64 + jl;
            prow[0]  = f2bf(e0 * rs);
            prow[16] = f2bf(e1 * rs);
            prow[32] = f2bf(e2 * rs);
            prow[48] = f2bf(e3 * rs);
        }
    }

    // ---- Phase 5: O_h = P_h [64,64] x V_h [64,32] (same wave, no barrier) ----
    for (int mt = 0; mt < 4; ++mt) {
        const unsigned short* arow = sP + h * 4096 + (mt * 16 + jl) * 64;
#pragma unroll
        for (int nt = 0; nt < 2; ++nt) {
            v8f acc = {};
            const unsigned short* bcol = sVT + h * 2048 + (nt * 16 + jl) * 64;
#pragma unroll
            for (int ks = 0; ks < 2; ++ks)
                acc = wmma_bf16(load_matA(arow + ks * 32, lane),
                                load_matB(bcol + ks * 32, lane), acc);
            const int j = h * 32 + nt * 16 + jl;      // [tok][h*32+d] layout
#pragma unroll
            for (int vi = 0; vi < 8; ++vi)
                sXO[(mt * 16 + rhalf + vi) * 256 + j] = f2bf(acc[vi]);
        }
    }
    __syncthreads();

    // ---- Phase 6: output projection [64,256] x [256,256] + bias -> global ----
    float* og = out + (size_t)blk * (64 * 256);
    for (int t = wv; t < 64; t += 8) {                // 4 x 16 output tiles
        const int mt = t >> 4, ntc = t & 15;
        const int j = ntc * 16 + jl;
        v8f acc;
        const float pb = proj_b[j];
#pragma unroll
        for (int vi = 0; vi < 8; ++vi) acc[vi] = pb;  // C preload = proj bias
        const unsigned short* arow = sXO + (mt * 16 + jl) * 256;
        const unsigned short* bcol = projT + (size_t)j * 256;
#pragma unroll
        for (int ks = 0; ks < 8; ++ks)
            acc = wmma_bf16(load_matA(arow + ks * 32, lane),
                            load_matB(bcol + ks * 32, lane), acc);
#pragma unroll
        for (int vi = 0; vi < 8; ++vi)
            og[(mt * 16 + rhalf + vi) * 256 + j] = acc[vi];
    }
}

// ---------------------------------------------------------------------------
extern "C" void kernel_launch(void* const* d_in, const int* in_sizes, int n_in,
                              void* d_out, int out_size, void* d_ws, size_t ws_size,
                              hipStream_t stream) {
    const float* x            = (const float*)d_in[0];
    const float* mask         = (const float*)d_in[1];
    const float* qkv_w        = (const float*)d_in[2];
    const float* q_bias       = (const float*)d_in[3];
    const float* v_bias       = (const float*)d_in[4];
    const float* logit_scale  = (const float*)d_in[5];
    const float* coords_table = (const float*)d_in[6];
    const float* mlp1_w       = (const float*)d_in[7];
    const float* mlp1_b       = (const float*)d_in[8];
    const float* mlp2_w       = (const float*)d_in[9];
    const float* proj_w       = (const float*)d_in[10];
    const float* proj_b       = (const float*)d_in[11];
    const int*   rel_idx      = (const int*)d_in[12];

    char* ws = (char*)d_ws;
    unsigned short* qkvT     = (unsigned short*)(ws);            // 768*256*2 = 384 KB
    unsigned short* projT    = (unsigned short*)(ws + 393216);   // 256*256*2 = 128 KB
    float*          rel_bias = (float*)         (ws + 524288);   // 8*64*64*4 = 128 KB

    convert_weights<<<1024, 256, 0, stream>>>(qkv_w, proj_w, qkvT, projT);
    cpb_bias<<<64, 64, 0, stream>>>(coords_table, mlp1_w, mlp1_b, mlp2_w,
                                    rel_idx, rel_bias);
    win_attn_fused<<<2048, 256, 0, stream>>>(x, mask, qkvT, q_bias, v_bias,
                                             logit_scale, rel_bias, projT,
                                             proj_b, (float*)d_out);
}
